// MultiHeadAttention_11922829213910
// MI455X (gfx1250) — compile-verified, matrix operand order checked by reading, not run on hardware
//
#include <hip/hip_runtime.h>

// Problem constants (from reference)
#define BB   2
#define SS   2048
#define DD   1024
#define HH   16
#define HD   64
#define ROWS (BB * SS)          // 4096 rows in the projection GEMMs

typedef _Float16 half_t;
typedef __attribute__((ext_vector_type(16))) _Float16 v16h;
typedef __attribute__((ext_vector_type(8)))  _Float16 v8h;
typedef __attribute__((ext_vector_type(8)))  float    v8f;

// ---------------------------------------------------------------------------
// Fragment helper: A-layout (16-bit A, 16x32, ISA 7.12.2): lane l (l16=l&15)
// holds row M=l16; halves a[0..7]=K(klo..klo+7), a[8..15]=K(16+klo..) with
// klo = (lane>=16 ? 8 : 0).  B uses the symmetric lane-per-column layout.
// Both reduce to: 8 contiguous halves at p, 8 contiguous halves at p+16.
// ---------------------------------------------------------------------------
__device__ __forceinline__ v16h load_frag_f16(const half_t* p) {
    v8h lo = *(const v8h*)(p);
    v8h hi = *(const v8h*)(p + 16);
    v16h r;
#pragma unroll
    for (int i = 0; i < 8; ++i) { r[i] = lo[i]; r[8 + i] = hi[i]; }
    return r;
}

__device__ __forceinline__ v8f wmma_f16(v16h a, v16h b, v8f c) {
    return __builtin_amdgcn_wmma_f32_16x16x32_f16(
        false, a, false, b, (short)0, c, false, false);
}

// ---------------------------------------------------------------------------
// One-shot f32 -> f16 conversion (memory bound; feeds all later kernels).
// ---------------------------------------------------------------------------
__global__ __launch_bounds__(256) void cvt_f32_to_f16_kernel(
    const float* __restrict__ src, half_t* __restrict__ dst, int n8)
{
    const int idx = blockIdx.x * blockDim.x + threadIdx.x;
    if (idx >= n8) return;
    const float4* s = (const float4*)src + (size_t)idx * 2;
    float4 x0 = s[0];
    float4 x1 = s[1];
    v8h o;
    o[0] = (_Float16)x0.x; o[1] = (_Float16)x0.y;
    o[2] = (_Float16)x0.z; o[3] = (_Float16)x0.w;
    o[4] = (_Float16)x1.x; o[5] = (_Float16)x1.y;
    o[6] = (_Float16)x1.z; o[7] = (_Float16)x1.w;
    *(v8h*)(dst + (size_t)idx * 8) = o;
}

// ---------------------------------------------------------------------------
// Projection GEMM, all-f16 operands: out[s,e] = sum_d Xh[s,d]*Wh[e,d] + b[e].
//   transpose_out == 0 : out layout [B,H,S,Hd]   (Q, K)
//   transpose_out == 1 : out layout [B,H,Hd,S]   (V -> contiguous PV B-frags)
// One wave per 16(row) x 64(col) strip; A reused across 4 N-tiles.
// Explicit ping-pong fragment buffers: stage-X WMMAs consume buffer X while
// buffer Y's loads (issued one stage earlier) are still in flight; refill of
// X happens right after its WMMAs and is pinned by a sched_barrier so the
// loads get ~4 WMMAs of latency cover.
// ---------------------------------------------------------------------------
__global__ __launch_bounds__(256) void proj_wmma_kernel(
    const half_t* __restrict__ Xh, const half_t* __restrict__ Wh,
    const float* __restrict__ bias, half_t* __restrict__ out,
    int transpose_out)
{
    const int wave = blockIdx.x * (blockDim.x >> 5) + (threadIdx.x >> 5);
    const int lane = threadIdx.x & 31;
    const int l16  = lane & 15;
    const int hi   = lane >> 4;
    const int klo  = hi * 8;

    const int ESTRIPS = DD / 64;
    const int srow0   = (wave / ESTRIPS) * 16;
    const int ecol0   = (wave % ESTRIPS) * 64;

    const half_t* xrow = Xh + (size_t)(srow0 + l16) * DD + klo;
    const half_t* wrow[4];
#pragma unroll
    for (int t = 0; t < 4; ++t)
        wrow[t] = Wh + (size_t)(ecol0 + t * 16 + l16) * DD + klo;

    v8f acc[4];
#pragma unroll
    for (int t = 0; t < 4; ++t) acc[t] = (v8f)(0.0f);

    const int NSTEP = DD / 32;   // 32 k-steps of 32

    // prime: step 0 -> buffer A, step 1 -> buffer B
    v16h aA = load_frag_f16(xrow);
    v16h bA[4];
#pragma unroll
    for (int t = 0; t < 4; ++t) bA[t] = load_frag_f16(wrow[t]);
    v16h aB = load_frag_f16(xrow + 32);
    v16h bB[4];
#pragma unroll
    for (int t = 0; t < 4; ++t) bB[t] = load_frag_f16(wrow[t] + 32);
    __builtin_amdgcn_sched_barrier(0);

#pragma unroll 3
    for (int s = 0; s < NSTEP - 2; s += 2) {
        // ---- stage A: compute step s, refill buffer A with step s+2 ----
#pragma unroll
        for (int t = 0; t < 4; ++t) acc[t] = wmma_f16(aA, bA[t], acc[t]);
        {
            const int ko = (s + 2) * 32;
            aA = load_frag_f16(xrow + ko);
#pragma unroll
            for (int t = 0; t < 4; ++t) bA[t] = load_frag_f16(wrow[t] + ko);
        }
        __builtin_amdgcn_sched_barrier(0);
        // ---- stage B: compute step s+1, refill buffer B with step s+3 ----
#pragma unroll
        for (int t = 0; t < 4; ++t) acc[t] = wmma_f16(aB, bB[t], acc[t]);
        {
            const int ko = (s + 3) * 32;
            aB = load_frag_f16(xrow + ko);
#pragma unroll
            for (int t = 0; t < 4; ++t) bB[t] = load_frag_f16(wrow[t] + ko);
        }
        __builtin_amdgcn_sched_barrier(0);
    }
    // epilogue: steps NSTEP-2, NSTEP-1 (already resident)
#pragma unroll
    for (int t = 0; t < 4; ++t) acc[t] = wmma_f16(aA, bA[t], acc[t]);
#pragma unroll
    for (int t = 0; t < 4; ++t) acc[t] = wmma_f16(aB, bB[t], acc[t]);

    // Bias + store. C layout: acc[t][i] = out[srow0 + i + 8*hi][ecol0+t*16+l16]
#pragma unroll
    for (int t = 0; t < 4; ++t) {
        const int e    = ecol0 + t * 16 + l16;
        const float bv = bias[e];
        const int h    = e >> 6;
        const int d    = e & 63;
#pragma unroll
        for (int i = 0; i < 8; ++i) {
            const int sg   = srow0 + i + 8 * hi;
            const int b_   = sg >> 11;
            const int sloc = sg & 2047;
            const float vfin = acc[t][i] + bv;
            size_t idx;
            if (transpose_out)
                idx = (((size_t)(b_ * HH + h) * HD + d) * SS) + sloc;
            else
                idx = (((size_t)(b_ * HH + h) * SS + sloc) * HD) + d;
            out[idx] = (_Float16)vfin;
        }
    }
}

// ---------------------------------------------------------------------------
// Flash attention, one wave per (b, h, 16-row query tile).
// Qp, Kp: f16 [B,H,S,Hd];  VpT: f16 [B,H,Hd,S];  out: f32 [B,S,D].
// Per 32-key chunk: all 4 K-frag and 4 V-frag loads are issued up front and
// pinned with a sched_barrier; the V loads are consumed only after the
// softmax VALU/trans section, which covers their latency.
// ---------------------------------------------------------------------------
__global__ __launch_bounds__(256) void attn_wmma_kernel(
    const half_t* __restrict__ Qp, const half_t* __restrict__ Kp,
    const half_t* __restrict__ VpT, float* __restrict__ out)
{
    __shared__ __align__(16) half_t pbuf[8][16 * 32];   // per-wave P scratch

    const int wid  = threadIdx.x >> 5;
    const int lane = threadIdx.x & 31;
    const int l16  = lane & 15;
    const int hi   = lane >> 4;
    const int klo  = hi * 8;

    const int unit = blockIdx.x * (blockDim.x >> 5) + wid;
    const int QTILES = SS / 16;
    const int bh = unit / QTILES;
    const int q0 = (unit % QTILES) * 16;

    const half_t* qb = Qp  + (size_t)bh * SS * HD;
    const half_t* kb = Kp  + (size_t)bh * SS * HD;
    const half_t* vt = VpT + (size_t)bh * HD * SS;

    const half_t* qrow = qb + (size_t)(q0 + l16) * HD;
    const v16h qa0 = load_frag_f16(qrow + 0  + klo);
    const v16h qa1 = load_frag_f16(qrow + 32 + klo);

    v8f oacc[4];
#pragma unroll
    for (int t = 0; t < 4; ++t) oacc[t] = (v8f)(0.0f);
    float m_run[8], l_run[8];
#pragma unroll
    for (int i = 0; i < 8; ++i) { m_run[i] = -INFINITY; l_run[i] = 0.0f; }

    const float scale = 0.125f;   // 1/sqrt(64)

    for (int k0 = 0; k0 < SS; k0 += 32) {
        if (k0 + 32 < SS) {
            __builtin_prefetch(kb + (size_t)(k0 + 32) * HD, 0, 1);
            __builtin_prefetch(vt + (size_t)l16 * SS + k0 + 32, 0, 1);
        }
        // ---- issue all K and V fragment loads for this chunk ----
        const half_t* kr0 = kb + (size_t)(k0 + 0  + l16) * HD + klo;
        const half_t* kr1 = kb + (size_t)(k0 + 16 + l16) * HD + klo;
        v16h kb00 = load_frag_f16(kr0 + 0);
        v16h kb01 = load_frag_f16(kr0 + 32);
        v16h kb10 = load_frag_f16(kr1 + 0);
        v16h kb11 = load_frag_f16(kr1 + 32);
        v16h vb[4];
#pragma unroll
        for (int t = 0; t < 4; ++t)
            vb[t] = load_frag_f16(vt + (size_t)(t * 16 + l16) * SS + k0 + klo);
        __builtin_amdgcn_sched_barrier(0);   // keep loads above the compute

        // ---- S = Q * K^T (two 16-key n-tiles, two 32-wide k-steps) ----
        v8f sac0 = (v8f)(0.0f), sac1 = (v8f)(0.0f);
        sac0 = wmma_f16(qa0, kb00, sac0);
        sac0 = wmma_f16(qa1, kb01, sac0);
        sac1 = wmma_f16(qa0, kb10, sac1);
        sac1 = wmma_f16(qa1, kb11, sac1);

        // ---- online softmax over the 32 new keys ----
        float alpha[8];
#pragma unroll
        for (int i = 0; i < 8; ++i) {
            float s0 = sac0[i] * scale;
            float s1 = sac1[i] * scale;
            float v = fmaxf(s0, s1);
            v = fmaxf(v, __shfl_xor(v, 8, 16));
            v = fmaxf(v, __shfl_xor(v, 4, 16));
            v = fmaxf(v, __shfl_xor(v, 2, 16));
            v = fmaxf(v, __shfl_xor(v, 1, 16));
            const float mn = fmaxf(m_run[i], v);
            const float al = __expf(m_run[i] - mn);
            const float p0 = __expf(s0 - mn);
            const float p1 = __expf(s1 - mn);
            float rs = p0 + p1;
            rs += __shfl_xor(rs, 8, 16);
            rs += __shfl_xor(rs, 4, 16);
            rs += __shfl_xor(rs, 2, 16);
            rs += __shfl_xor(rs, 1, 16);
            l_run[i] = l_run[i] * al + rs;
            m_run[i] = mn;
            alpha[i] = al;
            // scatter P (C layout) into LDS tile [row][key_local]
            const int prow = i + 8 * hi;
            pbuf[wid][prow * 32 + 0  + l16] = (_Float16)p0;
            pbuf[wid][prow * 32 + 16 + l16] = (_Float16)p1;
        }
#pragma unroll
        for (int t = 0; t < 4; ++t)
#pragma unroll
            for (int i = 0; i < 8; ++i) oacc[t][i] *= alpha[i];

        // LDS stores must land before the transposed reads (same wave).
        asm volatile("s_wait_dscnt 0x0" ::: "memory");

        // ---- O += P * V ----
        const v16h pa = load_frag_f16(&pbuf[wid][l16 * 32 + klo]);
#pragma unroll
        for (int t = 0; t < 4; ++t)
            oacc[t] = wmma_f16(pa, vb[t], oacc[t]);
    }

    // ---- normalize and store: out[b, q0+M, h*64 + t*16 + l16] ----
    const int b_ = bh / HH;
    const int h_ = bh % HH;
    float inv[8];
#pragma unroll
    for (int i = 0; i < 8; ++i) inv[i] = 1.0f / l_run[i];
#pragma unroll
    for (int t = 0; t < 4; ++t) {
        const int dcol = h_ * HD + t * 16 + l16;
#pragma unroll
        for (int i = 0; i < 8; ++i) {
            const int M = i + 8 * hi;
            out[((size_t)(b_ * SS + q0 + M)) * DD + dcol] = oacc[t][i] * inv[i];
        }
    }
}

// ---------------------------------------------------------------------------
// Host-side launch. Inputs: q,k,v,Wq,bq,Wk,bk,Wv,bv (all f32).
// Workspace (f16 halves):
//   [0)   qh,kh,vh    : 3 x 4M  (converted activations)
//   [12M) wqh,wkh,wvh : 3 x 1M  (converted weights)
//   [15M) Qp,Kp,VpT   : 3 x 4M  (projected heads)
// Total 27M halves = 54 MB.
// ---------------------------------------------------------------------------
extern "C" void kernel_launch(void* const* d_in, const int* in_sizes, int n_in,
                              void* d_out, int out_size, void* d_ws, size_t ws_size,
                              hipStream_t stream) {
    const float* q  = (const float*)d_in[0];
    const float* k  = (const float*)d_in[1];
    const float* v  = (const float*)d_in[2];
    const float* Wq = (const float*)d_in[3];
    const float* bq = (const float*)d_in[4];
    const float* Wk = (const float*)d_in[5];
    const float* bk = (const float*)d_in[6];
    const float* Wv = (const float*)d_in[7];
    const float* bv = (const float*)d_in[8];
    float* out = (float*)d_out;

    const size_t XN = (size_t)ROWS * DD;   // 4,194,304
    const size_t WN = (size_t)DD * DD;     // 1,048,576
    half_t* ws  = (half_t*)d_ws;
    half_t* qh  = ws;
    half_t* kh  = qh  + XN;
    half_t* vh  = kh  + XN;
    half_t* wqh = vh  + XN;
    half_t* wkh = wqh + WN;
    half_t* wvh = wkh + WN;
    half_t* Qp  = wvh + WN;
    half_t* Kp  = Qp  + XN;
    half_t* VpT = Kp  + XN;

    // 1) one-shot f32 -> f16 conversions (memory bound)
    const int xBlocks = (int)(XN / 8 / 256);   // 2048
    const int wBlocks = (int)(WN / 8 / 256);   // 512
    cvt_f32_to_f16_kernel<<<xBlocks, 256, 0, stream>>>(q,  qh,  (int)(XN / 8));
    cvt_f32_to_f16_kernel<<<xBlocks, 256, 0, stream>>>(k,  kh,  (int)(XN / 8));
    cvt_f32_to_f16_kernel<<<xBlocks, 256, 0, stream>>>(v,  vh,  (int)(XN / 8));
    cvt_f32_to_f16_kernel<<<wBlocks, 256, 0, stream>>>(Wq, wqh, (int)(WN / 8));
    cvt_f32_to_f16_kernel<<<wBlocks, 256, 0, stream>>>(Wk, wkh, (int)(WN / 8));
    cvt_f32_to_f16_kernel<<<wBlocks, 256, 0, stream>>>(Wv, wvh, (int)(WN / 8));

    // 2) projections: waves = (ROWS/16)*(DD/64) = 4096 -> 512 blocks x 8 waves
    const int projBlocks = (ROWS / 16) * (DD / 64) / 8;
    proj_wmma_kernel<<<projBlocks, 256, 0, stream>>>(qh, wqh, bq, Qp, 0);
    proj_wmma_kernel<<<projBlocks, 256, 0, stream>>>(kh, wkh, bk, Kp, 0);
    proj_wmma_kernel<<<projBlocks, 256, 0, stream>>>(vh, wvh, bv, VpT, 1);

    // 3) attention: units = B*H*(S/16) = 4096 -> 512 blocks x 8 waves
    const int attnBlocks = BB * HH * (SS / 16) / 8;
    attn_wmma_kernel<<<attnBlocks, 256, 0, stream>>>(Qp, Kp, VpT, out);
}
